// GAT_42116449304850
// MI455X (gfx1250) — matrix operand order repaired and checked by reading, not compiled, not run on hardware
//
#include <hip/hip_runtime.h>
#include <hip/hip_bf16.h>

typedef __attribute__((ext_vector_type(16))) _Float16 v16h;
typedef __attribute__((ext_vector_type(8)))  float    v8f;

#define NEG_SLOPE 0.2f
#define FMAP_NEG_INF 0x007fffffu   // monotonic-uint encoding of -inf

__device__ __forceinline__ float lrelu(float x) { return x > 0.f ? x : NEG_SLOPE * x; }
// order-preserving float -> uint mapping (for atomicMax-based segment max)
__device__ __forceinline__ unsigned fmapf(float f) {
  int i = __float_as_int(f);
  return (i < 0) ? ~(unsigned)i : ((unsigned)i | 0x80000000u);
}
__device__ __forceinline__ float funmapf(unsigned u) {
  int i = (u & 0x80000000u) ? (int)(u & 0x7fffffffu) : ~(int)u;
  return __int_as_float(i);
}

// ---------------------------------------------------------------------------
// Pack B[K x ldb] (f32 row-major) into WMMA B-fragment layout (f16), columns
// zero-padded to 16*Ntiles.  Fragment layout (ISA 7.12.2, 32x16 f16 B):
// lane n<16 -> column n, K kt*32+0..7/+16..23; lane n+16 -> same column, K +8..15/+24..31.
// Flat index: (((ntile*Kt + kt)*32 + lane)*16 + h) halves.
// ---------------------------------------------------------------------------
__global__ void pack_b_kernel(const float* __restrict__ B, int ldb, int colLimit,
                              int K, int Ntiles, _Float16* __restrict__ Bf) {
  int idx = blockIdx.x * blockDim.x + threadIdx.x;
  int Kt = K >> 5;
  int total = Ntiles * Kt * 32 * 16;
  if (idx >= total) return;
  int h     = idx & 15;
  int lane  = (idx >> 4) & 31;
  int rest  = idx >> 9;
  int kt    = rest % Kt;
  int ntile = rest / Kt;
  int col = ntile * 16 + (lane & 15);
  int k   = kt * 32 + ((lane >> 4) ? 8 : 0) + (h & 7) + ((h & 8) ? 16 : 0);
  float v = (col < colLimit) ? B[(long)k * ldb + col] : 0.f;
  Bf[idx] = (_Float16)v;
}

// ---------------------------------------------------------------------------
// WMMA GEMM: C[M x ldc] = op(A)[M x K] * B (+ bias).  4 waves per block, one
// wave per 16 x (16*NT) output strip; A fragments loaded as 4x b128 per
// k-step and reused across NT tiles; B fragments pre-packed (aligned v16h).
// ---------------------------------------------------------------------------
template <int NT>
__global__ void wmma_gemm_kernel(const float* __restrict__ A, int lda,
                                 const _Float16* __restrict__ Bf,
                                 const float* __restrict__ bias,
                                 float* __restrict__ C, int ldc,
                                 int K, int colLimit, int reluA, int Mtiles) {
  const int mtile = blockIdx.x * 4 + (threadIdx.x >> 5);
  if (mtile >= Mtiles) return;
  const int ng    = blockIdx.y;          // group of NT n-tiles
  const int lane  = threadIdx.x & 31;
  const int row0  = mtile * 16;
  const int Kt    = K >> 5;

  // A fragment (16x32 f16): lanes 0-15 carry K +0..7 / +16..23 of row lane&15,
  // lanes 16-31 carry K +8..15 / +24..31 of the same rows.
  const int mA    = row0 + (lane & 15);
  const int koffA = (lane >> 4) ? 8 : 0;
  const float* Arow = A + (long)mA * lda + koffA;

  v8f acc[NT];
#pragma unroll
  for (int t = 0; t < NT; ++t) acc[t] = (v8f){};

  for (int kt = 0; kt < Kt; ++kt) {
    const float4* p = (const float4*)(Arow + kt * 32);
    float4 f0 = p[0], f1 = p[1];   // K +0..7   -> halves 0..7
    float4 f2 = p[4], f3 = p[5];   // K +16..23 -> halves 8..15
    if (reluA) {
      f0.x = fmaxf(f0.x, 0.f); f0.y = fmaxf(f0.y, 0.f); f0.z = fmaxf(f0.z, 0.f); f0.w = fmaxf(f0.w, 0.f);
      f1.x = fmaxf(f1.x, 0.f); f1.y = fmaxf(f1.y, 0.f); f1.z = fmaxf(f1.z, 0.f); f1.w = fmaxf(f1.w, 0.f);
      f2.x = fmaxf(f2.x, 0.f); f2.y = fmaxf(f2.y, 0.f); f2.z = fmaxf(f2.z, 0.f); f2.w = fmaxf(f2.w, 0.f);
      f3.x = fmaxf(f3.x, 0.f); f3.y = fmaxf(f3.y, 0.f); f3.z = fmaxf(f3.z, 0.f); f3.w = fmaxf(f3.w, 0.f);
    }
    v16h a;
    a[0]  = (_Float16)f0.x; a[1]  = (_Float16)f0.y; a[2]  = (_Float16)f0.z; a[3]  = (_Float16)f0.w;
    a[4]  = (_Float16)f1.x; a[5]  = (_Float16)f1.y; a[6]  = (_Float16)f1.z; a[7]  = (_Float16)f1.w;
    a[8]  = (_Float16)f2.x; a[9]  = (_Float16)f2.y; a[10] = (_Float16)f2.z; a[11] = (_Float16)f2.w;
    a[12] = (_Float16)f3.x; a[13] = (_Float16)f3.y; a[14] = (_Float16)f3.z; a[15] = (_Float16)f3.w;
#pragma unroll
    for (int t = 0; t < NT; ++t) {
      const v16h b = *(const v16h*)(Bf + (((long)(ng * NT + t) * Kt + kt) * 32 + lane) * 16);
      acc[t] = __builtin_amdgcn_wmma_f32_16x16x32_f16(
          false, a, false, b, (short)0, acc[t], false, false);
    }
  }

  // C/D layout: VGPR r -> (M = r + 8*(lane>=16), N = lane&15)
  const int mbase = row0 + ((lane >> 4) ? 8 : 0);
  const int nlane = lane & 15;
#pragma unroll
  for (int t = 0; t < NT; ++t) {
    int nC = (ng * NT + t) * 16 + nlane;
    if (nC < colLimit) {
      float badd = bias ? bias[nC] : 0.f;
#pragma unroll
      for (int r = 0; r < 8; ++r)
        C[(long)(mbase + r) * ldc + nC] = acc[t][r] + badd;
    }
  }
}

// ---------------------------------------------------------------------------
// Per-(node, head) attention logits: aS[n,h] = <h[n,h,:], att_src[h,:]>, same
// for dst.  One wave per (n,h), 32 channels across lanes, shfl reduction.
// ---------------------------------------------------------------------------
__global__ void att_sums_kernel(const float* __restrict__ hmat, int H,
                                const float* __restrict__ att_src,
                                const float* __restrict__ att_dst,
                                float* __restrict__ aS, float* __restrict__ aD,
                                int NH) {
  int wid  = (int)((blockIdx.x * (long)blockDim.x + threadIdx.x) >> 5);
  int lane = threadIdx.x & 31;
  if (wid >= NH) return;
  int n = wid / H, hd = wid - n * H;
  float v = hmat[(long)n * (H * 32) + hd * 32 + lane];
  float s = v * att_src[hd * 32 + lane];
  float d = v * att_dst[hd * 32 + lane];
#pragma unroll
  for (int off = 16; off > 0; off >>= 1) {
    s += __shfl_down(s, off, 32);
    d += __shfl_down(d, off, 32);
  }
  if (lane == 0) { aS[wid] = s; aD[wid] = d; }
}

__global__ void fill_u32_kernel(unsigned* p, unsigned v, long n) {
  long i = blockIdx.x * (long)blockDim.x + threadIdx.x;
  if (i < n) p[i] = v;
}
__global__ void fill_f32_kernel(float* p, float v, long n) {
  long i = blockIdx.x * (long)blockDim.x + threadIdx.x;
  if (i < n) p[i] = v;
}
__global__ void init_bias_kernel(float* __restrict__ out,
                                 const float* __restrict__ bias, int ld, long total) {
  long i = blockIdx.x * (long)blockDim.x + threadIdx.x;
  if (i < total) out[i] = bias[i % ld];
}

// segment max over dst; one thread per edge, unrolled over heads.
// Self-loops folded in: e >= E -> src = dst = e - E.
template <int H>
__global__ void edge_max_kernel(const int* __restrict__ src, const int* __restrict__ dst,
                                int E, int N,
                                const float* __restrict__ aS, const float* __restrict__ aD,
                                unsigned* __restrict__ m) {
  int e = (int)(blockIdx.x * (long)blockDim.x + threadIdx.x);
  if (e >= E + N) return;
  int s, d;
  if (e < E) { s = src[e]; d = dst[e]; } else { s = d = e - E; }
#pragma unroll
  for (int hd = 0; hd < H; ++hd) {
    float alpha = lrelu(aS[(long)s * H + hd] + aD[(long)d * H + hd]);
    atomicMax(&m[(long)d * H + hd], fmapf(alpha));
  }
}

// segment sum of exp(alpha - max); one thread per edge.
template <int H>
__global__ void edge_denom_kernel(const int* __restrict__ src, const int* __restrict__ dst,
                                  int E, int N,
                                  const float* __restrict__ aS, const float* __restrict__ aD,
                                  const unsigned* __restrict__ m,
                                  float* __restrict__ denom) {
  int e = (int)(blockIdx.x * (long)blockDim.x + threadIdx.x);
  if (e >= E + N) return;
  int s, d;
  if (e < E) { s = src[e]; d = dst[e]; } else { s = d = e - E; }
#pragma unroll
  for (int hd = 0; hd < H; ++hd) {
    float alpha = lrelu(aS[(long)s * H + hd] + aD[(long)d * H + hd]);
    float ea = __expf(alpha - funmapf(m[(long)d * H + hd]));
    atomicAdd(&denom[(long)d * H + hd], ea);
  }
}

// out[dst, :] += w * h[src, :]  — one wave per edge, CH channels across lanes.
// Softmax weight computed once per head in lanes 0..H-1, shfl-broadcast.
// h and out are L2-resident (<=102MB each vs 192MB L2): gathers + f32 atomics
// stay on-chip; loads/atomics are fully coalesced (stride-1 across lanes).
template <int CH, int H>
__global__ void edge_scatter_kernel(const int* __restrict__ src, const int* __restrict__ dst,
                                    int E, int N,
                                    const float* __restrict__ aS, const float* __restrict__ aD,
                                    const unsigned* __restrict__ m,
                                    const float* __restrict__ denom,
                                    const float* __restrict__ hmat,
                                    float* __restrict__ out) {
  int wid  = (int)((blockIdx.x * (long)blockDim.x + threadIdx.x) >> 5);
  int lane = threadIdx.x & 31;
  int Etot = E + N;
  if (wid >= Etot) return;
  int s, d;
  if (wid < E) { s = src[wid]; d = dst[wid]; } else { s = d = wid - E; }

  float wlane = 0.f;
  if (lane < H) {
    float alpha = lrelu(aS[(long)s * H + lane] + aD[(long)d * H + lane]);
    float ea  = __expf(alpha - funmapf(m[(long)d * H + lane]));
    float den = fmaxf(denom[(long)d * H + lane], 1e-16f);
    wlane = ea / den;
  }
#pragma unroll
  for (int j = 0; j < CH / 32; ++j) {
    int ch = lane + 32 * j;          // head index = ch>>5 = j
    float w = __shfl(wlane, j, 32);
    atomicAdd(&out[(long)d * CH + ch], w * hmat[(long)s * CH + ch]);
  }
}

// ---------------------------------------------------------------------------
extern "C" void kernel_launch(void* const* d_in, const int* in_sizes, int n_in,
                              void* d_out, int out_size, void* d_ws, size_t ws_size,
                              hipStream_t stream) {
  const float* x0   = (const float*)d_in[0];   // [N,128]
  const float* W1   = (const float*)d_in[1];   // [128,256]
  const float* as1  = (const float*)d_in[2];   // [8,32]
  const float* ad1  = (const float*)d_in[3];   // [8,32]
  const float* b1   = (const float*)d_in[4];   // [256]
  const float* W2   = (const float*)d_in[5];   // [256,32]
  const float* as2  = (const float*)d_in[6];   // [1,32]
  const float* ad2  = (const float*)d_in[7];   // [1,32]
  const float* b2   = (const float*)d_in[8];   // [32]
  const float* linW = (const float*)d_in[9];   // [32,40]
  const float* linb = (const float*)d_in[10];  // [40]
  const int*   ei   = (const int*)d_in[11];    // [2,E] (int32: jax x64 is off)

  const int N  = in_sizes[0] / 128;            // 100000 (multiple of 16)
  const int E  = in_sizes[11] / 2;             // 1000000
  const int H1 = 8, HID = 32, IN = 128, OUT = 40;
  const int HC1 = H1 * HID;                    // 256
  const int* srcIdx = ei;
  const int* dstIdx = ei + E;

  // workspace layout (floats); layer-2 buffers reuse the h1 region
  float* ws   = (float*)d_ws;
  float* h1   = ws;                         // N*256
  float* out1 = ws + (long)N * 256;         // N*256
  float* aS1  = ws + (long)N * 512;         // N*8
  float* aD1  = aS1 + (long)N * 8;          // N*8
  unsigned* m1 = (unsigned*)(aD1 + (long)N * 8);   // N*8
  float* den1 = (float*)m1 + (long)N * 8;          // N*8
  _Float16* Bf1 = (_Float16*)(den1 + (long)N * 8); // 16*4*512 halves (64KB)
  _Float16* Bf2 = Bf1 + 16 * 4 * 512;              // 2*8*512 halves
  _Float16* Bf3 = Bf2 + 2 * 8 * 512;               // 3*1*512 halves
  float* h2   = ws;                         // N*32 (reuses h1 region)
  float* out2 = ws + (long)N * 32;          // N*32
  float* aS2  = ws + (long)N * 64;          // N
  float* aD2  = aS2 + N;                    // N
  unsigned* m2 = (unsigned*)(aD2 + N);      // N
  float* den2 = (float*)m2 + N;             // N
  float* y    = (float*)d_out;              // [N,40]

  const int Mt = N / 16;                    // 6250 M-tiles
  const int TB = 256;
  auto cdiv = [](long a, long b) { return (unsigned)((a + b - 1) / b); };
  const unsigned GemmGx = cdiv(Mt, 4);      // 4 waves (M-tiles) per block

  // ---- pack weights into f16 WMMA-B fragments ------------------------------
  pack_b_kernel<<<cdiv(16 * 4 * 512, TB), TB, 0, stream>>>(W1, HC1, HC1, IN, 16, Bf1);
  pack_b_kernel<<<cdiv(2 * 8 * 512, TB), TB, 0, stream>>>(W2, HID, HID, HC1, 2, Bf2);
  pack_b_kernel<<<cdiv(3 * 1 * 512, TB), TB, 0, stream>>>(linW, OUT, OUT, HID, 3, Bf3);

  // ---- layer 1 -------------------------------------------------------------
  wmma_gemm_kernel<4><<<dim3(GemmGx, 4), 128, 0, stream>>>(
      x0, IN, Bf1, nullptr, h1, HC1, IN, HC1, 0, Mt);
  att_sums_kernel<<<cdiv((long)N * H1, 8), TB, 0, stream>>>(
      h1, H1, as1, ad1, aS1, aD1, N * H1);
  fill_u32_kernel<<<cdiv((long)N * H1, TB), TB, 0, stream>>>(m1, FMAP_NEG_INF, (long)N * H1);
  fill_f32_kernel<<<cdiv((long)N * H1, TB), TB, 0, stream>>>(den1, 0.f, (long)N * H1);
  edge_max_kernel<8><<<cdiv((long)(E + N), TB), TB, 0, stream>>>(
      srcIdx, dstIdx, E, N, aS1, aD1, m1);
  edge_denom_kernel<8><<<cdiv((long)(E + N), TB), TB, 0, stream>>>(
      srcIdx, dstIdx, E, N, aS1, aD1, m1, den1);
  init_bias_kernel<<<cdiv((long)N * HC1, TB), TB, 0, stream>>>(out1, b1, HC1, (long)N * HC1);
  edge_scatter_kernel<256, 8><<<cdiv((long)(E + N) * 32, TB), TB, 0, stream>>>(
      srcIdx, dstIdx, E, N, aS1, aD1, m1, den1, h1, out1);

  // ---- layer 2 (x2 = relu(out1)) ------------------------------------------
  wmma_gemm_kernel<2><<<dim3(GemmGx, 1), 128, 0, stream>>>(
      out1, HC1, Bf2, nullptr, h2, HID, HC1, HID, /*reluA=*/1, Mt);
  att_sums_kernel<<<cdiv((long)N, 8), TB, 0, stream>>>(h2, 1, as2, ad2, aS2, aD2, N);
  fill_u32_kernel<<<cdiv((long)N, TB), TB, 0, stream>>>(m2, FMAP_NEG_INF, N);
  fill_f32_kernel<<<cdiv((long)N, TB), TB, 0, stream>>>(den2, 0.f, N);
  edge_max_kernel<1><<<cdiv((long)(E + N), TB), TB, 0, stream>>>(
      srcIdx, dstIdx, E, N, aS2, aD2, m2);
  edge_denom_kernel<1><<<cdiv((long)(E + N), TB), TB, 0, stream>>>(
      srcIdx, dstIdx, E, N, aS2, aD2, m2, den2);
  init_bias_kernel<<<cdiv((long)N * HID, TB), TB, 0, stream>>>(out2, b2, HID, (long)N * HID);
  edge_scatter_kernel<32, 1><<<cdiv((long)(E + N) * 32, TB), TB, 0, stream>>>(
      srcIdx, dstIdx, E, N, aS2, aD2, m2, den2, h2, out2);

  // ---- final linear: y = out2 @ linW + linb (40 cols -> 3 tiles) ----------
  wmma_gemm_kernel<3><<<dim3(GemmGx, 1), 128, 0, stream>>>(
      out2, HID, Bf3, linb, y, OUT, HID, OUT, 0, Mt);

  (void)n_in; (void)out_size; (void)ws_size;
}
